// Transformer_19765439496488
// MI455X (gfx1250) — compile-verified, matrix operand order checked by reading, not run on hardware
//
#include <hip/hip_runtime.h>
#include <hip/hip_bf16.h>
#include <math.h>

// ---------------------------------------------------------------------------
// Types for CDNA5 WMMA
// ---------------------------------------------------------------------------
typedef __attribute__((ext_vector_type(16))) __bf16 v16bf;
typedef __attribute__((ext_vector_type(2)))  __bf16 v2bf;
typedef __attribute__((ext_vector_type(8)))  float  v8f;

union BFrag { v16bf v; unsigned u[8]; };

__device__ __forceinline__ unsigned pack2(float lo, float hi) {
#if __has_builtin(__builtin_amdgcn_cvt_pk_bf16_f32)
    union { v2bf v; unsigned u; } cv;
    cv.v = __builtin_amdgcn_cvt_pk_bf16_f32(lo, hi);   // v_cvt_pk_bf16_f32
    return cv.u;
#else
    // native fptrunc (RNE); backend selects hardware bf16 cvt if available
    union { v2bf v; unsigned u; } cv;
    cv.v = (v2bf){(__bf16)lo, (__bf16)hi};
    return cv.u;
#endif
}

// ---------------------------------------------------------------------------
// Batched GEMM: C[z] = A[z] @ B[z] (+bias)(+relu), fp32 in/out, bf16 WMMA core
//   A: M x K, row stride lda
//   B: if transB: (N x K) row stride ldb (already B^T, e.g. K for QK^T)
//      else     : (K x N) row stride ldb (row-major weight)
//   C: M x N, row stride ldc
//   batch z = blockIdx.z; b = z/Hdim, h = z%Hdim
// Block: 256 threads (8 wave32). Tile 128x64, K-step 32, double-buffered LDS.
// LDS tiles stored as packed bf16x2 dwords: tile[row][kpair].
// Requires: M%128==0, N%64==0, K%32==0 (true for every shape in this model)
// ---------------------------------------------------------------------------
#define BM 128
#define BN 64
#define BKK 32
#define KP  (BKK / 2)   // 16 k-pairs

__global__ __launch_bounds__(256) void gemm_bf16_kernel(
    const float* __restrict__ A, const float* __restrict__ B,
    float* __restrict__ C, const float* __restrict__ bias,
    int M, int N, int K, int lda, int ldb, int ldc,
    int transB, int relu, int Hdim,
    long sAb, long sAh, long sBb, long sBh, long sCb, long sCh)
{
    __shared__ alignas(16) unsigned Asm[2][BM * KP];   // [m][kp] packed bf16x2
    __shared__ alignas(16) unsigned Bsm[2][BN * KP];   // Bt: [n][kp] packed bf16x2

    const int z  = blockIdx.z;
    const int bb = z / Hdim;
    const int hh = z % Hdim;
    const float* Ap = A + bb * sAb + hh * sAh;
    const float* Bp = B + bb * sBb + hh * sBh;
    float*       Cp = C + bb * sCb + hh * sCh;

    const int tid   = threadIdx.x;
    const int tileM = blockIdx.y * BM;
    const int tileN = blockIdx.x * BN;

    const int wave = tid >> 5;
    const int lane = tid & 31;
    const int wm = wave >> 1;       // 0..3  (rows of 32)
    const int wn = wave & 1;        // 0..1  (cols of 32)
    const int lm = lane & 15;
    const int lh = lane >> 4;

    v8f acc[2][2] = {};
    int buf = 0;

    for (int k0 = 0; k0 < K; k0 += BKK) {
        // ---- stage A tile: 128x32 fp32 -> packed bf16x2 (float2 loads) ----
        #pragma unroll
        for (int it = 0; it < (BM * KP) / 256; ++it) {      // 8 iters
            int idx = tid + it * 256;
            int m = idx >> 4, kp = idx & 15;
            float2 f = *(const float2*)(Ap + (long)(tileM + m) * lda + k0 + 2 * kp);
            Asm[buf][idx] = pack2(f.x, f.y);
        }
        // ---- stage B tile as Bt[n][kp] ----
        if (transB) {
            #pragma unroll
            for (int it = 0; it < (BN * KP) / 256; ++it) {  // 4 iters
                int idx = tid + it * 256;
                int n = idx >> 4, kp = idx & 15;
                float2 f = *(const float2*)(Bp + (long)(tileN + n) * ldb + k0 + 2 * kp);
                Bsm[buf][idx] = pack2(f.x, f.y);
            }
        } else {
            #pragma unroll
            for (int it = 0; it < (BN * KP) / 256; ++it) {  // n-fastest: coalesced
                int idx = tid + it * 256;
                int n = idx & 63, kp = idx >> 6;
                float f0 = Bp[(long)(k0 + 2 * kp)     * ldb + tileN + n];
                float f1 = Bp[(long)(k0 + 2 * kp + 1) * ldb + tileN + n];
                Bsm[buf][n * KP + kp] = pack2(f0, f1);
            }
        }

        // ---- prefetch next K-tile while this one is consumed ----
        if (k0 + BKK < K) {
            int pm = tid >> 1;
            __builtin_prefetch(Ap + (long)(tileM + pm) * lda + k0 + BKK + (tid & 1) * 16, 0, 0);
            if (transB) {
                int pn = tid & 63;
                __builtin_prefetch(Bp + (long)(tileN + pn) * ldb + k0 + BKK + (tid >> 6) * 8, 0, 0);
            } else {
                int pk = tid >> 3;
                __builtin_prefetch(Bp + (long)(k0 + BKK + pk) * ldb + tileN + (tid & 7) * 8, 0, 0);
            }
        }

        __syncthreads();

        // ---- fragments per ISA 16-bit A 16x32 layout ----
        //   lane: m = lane&15, half = lane>>4
        //   VGPR v: kpair = (v&3) + half*4 + (v>=4 ? 8 : 0)
        BFrag af[2], bf[2];
        #pragma unroll
        for (int t = 0; t < 2; ++t) {
            int m0 = wm * 32 + t * 16 + lm;
            int n0 = wn * 32 + t * 16 + lm;
            #pragma unroll
            for (int v = 0; v < 8; ++v) {
                int kp = (v & 3) + lh * 4 + ((v & 4) << 1);
                af[t].u[v] = Asm[buf][m0 * KP + kp];
                bf[t].u[v] = Bsm[buf][n0 * KP + kp];
            }
        }

        #pragma unroll
        for (int tm = 0; tm < 2; ++tm)
            #pragma unroll
            for (int tn = 0; tn < 2; ++tn)
                acc[tm][tn] = __builtin_amdgcn_wmma_f32_16x16x32_bf16(
                    false, af[tm].v, false, bf[tn].v,
                    (short)0, acc[tm][tn], false, false);

        buf ^= 1;
        __syncthreads();
    }

    // ---- epilogue: bias (preloaded, uniform), optional relu, store ----
    float bv[2] = {0.0f, 0.0f};
    if (bias) {
        bv[0] = bias[tileN + wn * 32 + lm];
        bv[1] = bias[tileN + wn * 32 + 16 + lm];
    }
    #pragma unroll
    for (int tm = 0; tm < 2; ++tm)
        #pragma unroll
        for (int tn = 0; tn < 2; ++tn)
            #pragma unroll
            for (int v = 0; v < 8; ++v)
                acc[tm][tn][v] += bv[tn];
    if (relu) {
        #pragma unroll
        for (int tm = 0; tm < 2; ++tm)
            #pragma unroll
            for (int tn = 0; tn < 2; ++tn)
                #pragma unroll
                for (int v = 0; v < 8; ++v)
                    acc[tm][tn][v] = fmaxf(acc[tm][tn][v], 0.0f);
    }
    #pragma unroll
    for (int tm = 0; tm < 2; ++tm)
        #pragma unroll
        for (int tn = 0; tn < 2; ++tn)
            #pragma unroll
            for (int v = 0; v < 8; ++v) {
                int m = tileM + wm * 32 + tm * 16 + lh * 8 + v;
                int n = tileN + wn * 32 + tn * 16 + lm;
                Cp[(long)m * ldc + n] = acc[tm][tn][v];
            }
}

// ---------------------------------------------------------------------------
// Row softmax with scale and optional causal mask (col > row%qlen -> 0)
// ---------------------------------------------------------------------------
__global__ __launch_bounds__(256) void softmax_kernel(
    float* __restrict__ sc, int cols, int qlen, float scale, int causal)
{
    __shared__ float red[256];
    const int row = blockIdx.x;
    const int q   = row % qlen;
    float* p = sc + (size_t)row * cols;
    const int tid = threadIdx.x;

    float mx = -3.0e38f;
    for (int c = tid; c < cols; c += 256) {
        float v = (causal && c > q) ? -3.0e38f : p[c] * scale;
        mx = fmaxf(mx, v);
    }
    red[tid] = mx; __syncthreads();
    for (int s = 128; s > 0; s >>= 1) {
        if (tid < s) red[tid] = fmaxf(red[tid], red[tid + s]);
        __syncthreads();
    }
    mx = red[0]; __syncthreads();

    float sum = 0.0f;
    for (int c = tid; c < cols; c += 256) {
        float v = (causal && c > q) ? 0.0f : __expf(p[c] * scale - mx);
        p[c] = v;
        sum += v;
    }
    red[tid] = sum; __syncthreads();
    for (int s = 128; s > 0; s >>= 1) {
        if (tid < s) red[tid] += red[tid + s];
        __syncthreads();
    }
    float inv = 1.0f / red[0];
    for (int c = tid; c < cols; c += 256) p[c] *= inv;
}

// ---------------------------------------------------------------------------
// out = LayerNorm(x + r) * w + b   (one block per row; out may alias x)
// ---------------------------------------------------------------------------
__global__ __launch_bounds__(256) void ln_kernel(
    const float* __restrict__ x, const float* __restrict__ r,
    const float* __restrict__ w, const float* __restrict__ b,
    float* __restrict__ out, int cols)
{
    __shared__ float r1[256], r2[256];
    const int row = blockIdx.x;
    const int tid = threadIdx.x;
    const float* xp = x + (size_t)row * cols;
    const float* rp = r + (size_t)row * cols;
    float* op = out + (size_t)row * cols;

    float s = 0.0f, s2 = 0.0f;
    for (int c = tid; c < cols; c += 256) {
        float v = xp[c] + rp[c];
        s += v; s2 += v * v;
    }
    r1[tid] = s; r2[tid] = s2; __syncthreads();
    for (int st = 128; st > 0; st >>= 1) {
        if (tid < st) { r1[tid] += r1[tid + st]; r2[tid] += r2[tid + st]; }
        __syncthreads();
    }
    float mean = r1[0] / cols;
    float var  = r2[0] / cols - mean * mean;
    float rstd = rsqrtf(var + 1e-5f);
    __syncthreads();
    for (int c = tid; c < cols; c += 256) {
        float v = xp[c] + rp[c];
        op[c] = (v - mean) * rstd * w[c] + b[c];
    }
}

// ---------------------------------------------------------------------------
// out[row, d] = emb[tok[row], d] * scale + PE(row%seqlen, d)
// ---------------------------------------------------------------------------
__global__ __launch_bounds__(256) void embed_kernel(
    const int* __restrict__ tok, const float* __restrict__ emb,
    float* __restrict__ out, int seqlen, int dmodel, float scale)
{
    const int row = blockIdx.x;
    const int pos = row % seqlen;
    const int t   = tok[row];
    const float nl = -9.210340371976184f / (float)dmodel;  // -ln(10000)/D
    for (int d = threadIdx.x; d < dmodel; d += 256) {
        int i2 = d & ~1;
        float dv  = __expf((float)i2 * nl);
        float ang = (float)pos * dv;
        float pe  = (d & 1) ? __cosf(ang) : __sinf(ang);
        out[(size_t)row * dmodel + d] = emb[(size_t)t * dmodel + d] * scale + pe;
    }
}

// ---------------------------------------------------------------------------
// Host orchestration
// ---------------------------------------------------------------------------
static void launch_gemm(hipStream_t st, const float* A, const float* B, float* C,
                        const float* bias, int M, int N, int K,
                        int lda, int ldb, int ldc, bool transB, bool relu,
                        int batch = 1, int Hdim = 1,
                        long sAb = 0, long sAh = 0, long sBb = 0, long sBh = 0,
                        long sCb = 0, long sCh = 0)
{
    dim3 grid(N / BN, M / BM, batch);
    gemm_bf16_kernel<<<grid, 256, 0, st>>>(A, B, C, bias, M, N, K, lda, ldb, ldc,
                                           transB ? 1 : 0, relu ? 1 : 0, Hdim,
                                           sAb, sAh, sBb, sBh, sCb, sCh);
}

extern "C" void kernel_launch(void* const* d_in, const int* in_sizes, int n_in,
                              void* d_out, int out_size, void* d_ws, size_t ws_size,
                              hipStream_t stream) {
    (void)in_sizes; (void)n_in; (void)out_size; (void)ws_size;

    const int Dm = 512, Hh = 8, FFd = 2048, Ll = 6, Bb = 4, Ss = 512, Tt = 512, DKk = 64;
    const int Rows = Bb * Ss;                 // 2048 rows (encoder == decoder length)
    const float scale_emb  = sqrtf((float)Dm);
    const float scale_attn = 1.0f / sqrtf((float)DKk);

    // ---- input pointers (setup_inputs dict order, nested dicts flattened) ----
    int i = 0;
    const int*   src      = (const int*)  d_in[i++];  // 0
    const int*   tgt      = (const int*)  d_in[i++];  // 1
    const float* src_emb  = (const float*)d_in[i++];  // 2
    const float* tgt_emb  = (const float*)d_in[i++];  // 3
    const float* eWq      = (const float*)d_in[i++];  // 4
    const float* eWk      = (const float*)d_in[i++];
    const float* eWv      = (const float*)d_in[i++];
    const float* eWo      = (const float*)d_in[i++];
    const float* eln1w    = (const float*)d_in[i++];
    const float* eln1b    = (const float*)d_in[i++];
    const float* eln2w    = (const float*)d_in[i++];
    const float* eln2b    = (const float*)d_in[i++];
    const float* eW1      = (const float*)d_in[i++];
    const float* eb1      = (const float*)d_in[i++];
    const float* eW2      = (const float*)d_in[i++];
    const float* eb2      = (const float*)d_in[i++];
    const float* dsWq     = (const float*)d_in[i++];  // 16
    const float* dsWk     = (const float*)d_in[i++];
    const float* dsWv     = (const float*)d_in[i++];
    const float* dsWo     = (const float*)d_in[i++];
    const float* dcWq     = (const float*)d_in[i++];
    const float* dcWk     = (const float*)d_in[i++];
    const float* dcWv     = (const float*)d_in[i++];
    const float* dcWo     = (const float*)d_in[i++];
    const float* dln1w    = (const float*)d_in[i++];
    const float* dln1b    = (const float*)d_in[i++];
    const float* dln2w    = (const float*)d_in[i++];
    const float* dln2b    = (const float*)d_in[i++];
    const float* dln3w    = (const float*)d_in[i++];
    const float* dln3b    = (const float*)d_in[i++];
    const float* dW1      = (const float*)d_in[i++];
    const float* db1      = (const float*)d_in[i++];
    const float* dW2      = (const float*)d_in[i++];
    const float* db2      = (const float*)d_in[i++];
    const float* out_W    = (const float*)d_in[i++];  // 34
    const float* out_b    = (const float*)d_in[i++];  // 35

    // ---- workspace carve (floats) ----
    float* ws   = (float*)d_ws;
    const long NXD = (long)Rows * Dm;                 // 1,048,576
    float* bx   = ws;                                 // activations (x, then y)
    float* benc = bx   + NXD;                         // saved encoder output
    float* bq   = benc + NXD;
    float* bk   = bq   + NXD;
    float* bv   = bk   + NXD;
    float* bt   = bv   + NXD;                         // attention context
    float* ba   = bt   + NXD;                         // sublayer output
    float* bsc  = ba   + NXD;                         // scores: B*H*S*S = 8,388,608
    float* bff  = bsc + (long)Bb * Hh * Ss * Ss;      // FFN hidden: 4,194,304

    const long sXb = (long)Ss * Dm;                   // per-batch activation stride
    const long sSc = (long)Ss * Ss;                   // per-head score stride

    // one full attention block (q from X, k/v from KV)
    auto attention = [&](const float* X, const float* KV,
                         const float* Wq, const float* Wk, const float* Wv,
                         const float* Wo, int causal) {
        launch_gemm(stream, X,  Wq, bq, nullptr, Rows, Dm, Dm, Dm, Dm, Dm, false, false);
        launch_gemm(stream, KV, Wk, bk, nullptr, Rows, Dm, Dm, Dm, Dm, Dm, false, false);
        launch_gemm(stream, KV, Wv, bv, nullptr, Rows, Dm, Dm, Dm, Dm, Dm, false, false);
        // scores[z] = Q[b,:,h,:] @ K[b,:,h,:]^T   (transB: K rows are N x K)
        launch_gemm(stream, bq, bk, bsc, nullptr, Ss, Ss, DKk, Dm, Dm, Ss,
                    true, false, Bb * Hh, Hh, sXb, DKk, sXb, DKk,
                    (long)Hh * sSc, sSc);
        softmax_kernel<<<Bb * Hh * Ss, 256, 0, stream>>>(bsc, Ss, Ss, scale_attn, causal);
        // ctx[b,:,h,:] = P @ V[b,:,h,:]
        launch_gemm(stream, bsc, bv, bt, nullptr, Ss, DKk, Ss, Ss, Dm, Dm,
                    false, false, Bb * Hh, Hh, (long)Hh * sSc, sSc,
                    sXb, DKk, sXb, DKk);
        launch_gemm(stream, bt, Wo, ba, nullptr, Rows, Dm, Dm, Dm, Dm, Dm, false, false);
    };

    // ================= encoder =================
    embed_kernel<<<Rows, 256, 0, stream>>>(src, src_emb, bx, Ss, Dm, scale_emb);
    for (int l = 0; l < Ll; ++l) {
        const long oW = (long)l * Dm * Dm, oD = (long)l * Dm;
        const long o1 = (long)l * Dm * FFd, oF = (long)l * FFd;
        attention(bx, bx, eWq + oW, eWk + oW, eWv + oW, eWo + oW, 0);
        ln_kernel<<<Rows, 256, 0, stream>>>(bx, ba, eln1w + oD, eln1b + oD, bx, Dm);
        launch_gemm(stream, bx,  eW1 + o1, bff, eb1 + oF, Rows, FFd, Dm, Dm, FFd, FFd, false, true);
        launch_gemm(stream, bff, eW2 + o1, ba,  eb2 + oD, Rows, Dm, FFd, FFd, Dm, Dm, false, false);
        ln_kernel<<<Rows, 256, 0, stream>>>(bx, ba, eln2w + oD, eln2b + oD, bx, Dm);
    }
    hipMemcpyAsync(benc, bx, (size_t)NXD * sizeof(float),
                   hipMemcpyDeviceToDevice, stream);

    // ================= decoder =================
    embed_kernel<<<Rows, 256, 0, stream>>>(tgt, tgt_emb, bx, Tt, Dm, scale_emb);
    for (int l = 0; l < Ll; ++l) {
        const long oW = (long)l * Dm * Dm, oD = (long)l * Dm;
        const long o1 = (long)l * Dm * FFd, oF = (long)l * FFd;
        // masked self-attention
        attention(bx, bx, dsWq + oW, dsWk + oW, dsWv + oW, dsWo + oW, 1);
        ln_kernel<<<Rows, 256, 0, stream>>>(bx, ba, dln1w + oD, dln1b + oD, bx, Dm);
        // cross-attention over encoder output
        attention(bx, benc, dcWq + oW, dcWk + oW, dcWv + oW, dcWo + oW, 0);
        ln_kernel<<<Rows, 256, 0, stream>>>(bx, ba, dln2w + oD, dln2b + oD, bx, Dm);
        // FFN
        launch_gemm(stream, bx,  dW1 + o1, bff, db1 + oF, Rows, FFd, Dm, Dm, FFd, FFd, false, true);
        launch_gemm(stream, bff, dW2 + o1, ba,  db2 + oD, Rows, Dm, FFd, FFd, Dm, Dm, false, false);
        ln_kernel<<<Rows, 256, 0, stream>>>(bx, ba, dln3w + oD, dln3b + oD, bx, Dm);
    }

    // ================= output projection =================
    const int Vv = 32000;
    launch_gemm(stream, bx, out_W, (float*)d_out, out_b,
                Rows, Vv, Dm, Dm, Vv, Vv, false, false);
}
